// FHDF2d_695784702115
// MI455X (gfx1250) — matrix operand order repaired and checked by reading, not compiled, not run on hardware
//
#include <hip/hip_runtime.h>
#include <stdint.h>

#define B_  4
#define F_  8
#define K_  16
#define H_  512
#define W_  512
#define N_  (H_*W_)
#define RAD 7
#define KS  15

typedef __attribute__((ext_vector_type(16))) _Float16 v16h;
typedef __attribute__((ext_vector_type(8)))  float    v8f;

// spatial kernel exp(-r^2/4), r = -7..7 (unnormalized, matches reference)
__device__ const float SW[KS] = {
    4.7851173e-06f, 1.2340980e-04f, 1.9304541e-03f, 1.8315639e-02f,
    1.0539922e-01f, 3.6787944e-01f, 7.7880078e-01f, 1.0f,
    7.7880078e-01f, 3.6787944e-01f, 1.0539922e-01f, 1.8315639e-02f,
    1.9304541e-03f, 1.2340980e-04f, 4.7851173e-06f };

// ---------------- pack guide (B,8,H,W) f32 -> (B,N,8) f16 + |g|^2 ----------------
__global__ void __launch_bounds__(256) pack_guide(const float* __restrict__ guide,
                                                  _Float16* __restrict__ gp,
                                                  float* __restrict__ gsq) {
  int gid = blockIdx.x * 256 + threadIdx.x;         // 0 .. B*N-1
  int b = gid >> 18;
  int n = gid & (N_ - 1);
  union { uint4 u; _Float16 h[8]; } cv;
  float s = 0.f;
#pragma unroll
  for (int f = 0; f < F_; ++f) {
    float v = guide[((size_t)b * F_ + f) * N_ + n];
    _Float16 hv = (_Float16)v;
    cv.h[f] = hv;
    float vf = (float)hv;
    s += vf * vf;
  }
  *reinterpret_cast<uint4*>(gp + (size_t)gid * 8) = cv.u;
  gsq[gid] = s;
}

// ---------------- centers0 = first 16 pixels ----------------
__global__ void init_centers(const _Float16* __restrict__ gp, float* __restrict__ centers) {
  int b = blockIdx.x, tid = threadIdx.x;            // 128 threads: (k,f)
  int k = tid >> 3, f = tid & 7;
  centers[(b * K_ + k) * F_ + f] = (float)gp[((size_t)b * N_ + k) * 8 + f];
}

// ---------------- kmeans: assignment + accumulation ----------------
__global__ void __launch_bounds__(256) assign_accum(const _Float16* __restrict__ gp,
                                                    const float* __restrict__ centers,
                                                    float* __restrict__ sums,
                                                    float* __restrict__ counts) {
  __shared__ float c_l[K_ * F_];
  __shared__ float csq_l[K_];
  __shared__ float s_l[K_ * F_];
  __shared__ float cnt_l[K_];
  int b  = blockIdx.x >> 7;                          // 128 blocks per batch
  int bb = blockIdx.x & 127;
  int tid = threadIdx.x;
  if (tid < K_ * F_) { c_l[tid] = centers[b * K_ * F_ + tid]; s_l[tid] = 0.f; }
  if (tid < K_) cnt_l[tid] = 0.f;
  __syncthreads();
  if (tid < K_) {
    float s = 0.f;
#pragma unroll
    for (int f = 0; f < F_; ++f) { float c = c_l[tid * F_ + f]; s += c * c; }
    csq_l[tid] = s;
  }
  __syncthreads();
#pragma unroll 1
  for (int j = 0; j < 8; ++j) {
    int n = bb * 2048 + j * 256 + tid;
    union { uint4 u; _Float16 h[8]; } cv;
    cv.u = *reinterpret_cast<const uint4*>(gp + ((size_t)b * N_ + n) * 8);
    float fv[F_];
#pragma unroll
    for (int f = 0; f < F_; ++f) fv[f] = (float)cv.h[f];
    float best = 3.4e38f; int bi = 0;
#pragma unroll
    for (int k = 0; k < K_; ++k) {
      float dot = 0.f;
#pragma unroll
      for (int f = 0; f < F_; ++f) dot += fv[f] * c_l[k * F_ + f];
      float d = csq_l[k] - 2.f * dot;                // argmin (gsq constant per pixel)
      if (d < best) { best = d; bi = k; }
    }
#pragma unroll
    for (int f = 0; f < F_; ++f) atomicAdd(&s_l[bi * F_ + f], fv[f]);
    atomicAdd(&cnt_l[bi], 1.f);
  }
  __syncthreads();
  if (tid < K_ * F_) atomicAdd(&sums[b * K_ * F_ + tid], s_l[tid]);
  if (tid < K_) atomicAdd(&counts[b * K_ + tid], cnt_l[tid]);
}

// ---------------- kmeans: center update ----------------
__global__ void update_centers(float* __restrict__ centers,
                               const float* __restrict__ sums,
                               const float* __restrict__ counts) {
  int b = blockIdx.x, tid = threadIdx.x;             // 128 threads
  int k = tid >> 3;
  float cnt  = counts[b * K_ + k];
  float s    = sums[b * K_ * F_ + tid];
  float oldc = centers[b * K_ * F_ + tid];
  float nc   = s / fmaxf(cnt, 1.f);
  centers[b * K_ * F_ + tid] = (cnt > 0.5f) ? nc : oldc;
}

// ---------------- A = exp(-|ci-cj|^2/4); Ad ~ pinv(A) via ridge Gauss-Jordan ----------------
__global__ void pinv16(const float* __restrict__ centers, float* __restrict__ Ad) {
  __shared__ float M[16][33];
  __shared__ int piv;
  int b = blockIdx.x, tid = threadIdx.x;             // one wave (32 threads)
  for (int idx = tid; idx < 256; idx += 32) {
    int i = idx >> 4, j = idx & 15;
    float d = 0.f;
#pragma unroll
    for (int f = 0; f < F_; ++f) {
      float df = centers[(b * K_ + i) * F_ + f] - centers[(b * K_ + j) * F_ + f];
      d += df * df;
    }
    M[i][j] = __expf(-d * 0.25f) + ((i == j) ? 1e-5f : 0.f);
    M[i][16 + j] = (i == j) ? 1.f : 0.f;
  }
  __syncthreads();
  for (int col = 0; col < 16; ++col) {
    if (tid == 0) {
      int p = col; float mx = fabsf(M[col][col]);
      for (int r = col + 1; r < 16; ++r) { float v = fabsf(M[r][col]); if (v > mx) { mx = v; p = r; } }
      piv = p;
    }
    __syncthreads();
    int p = piv;
    if (p != col) { float t = M[col][tid]; M[col][tid] = M[p][tid]; M[p][tid] = t; }
    __syncthreads();
    float pv = M[col][col];
    M[col][tid] = M[col][tid] / pv;
    __syncthreads();
    for (int r = 0; r < 16; ++r) {
      if (r == col) continue;
      float f = M[r][col];                           // lockstep wave: load precedes stores
      M[r][tid] -= f * M[col][tid];
    }
    __syncthreads();
  }
  for (int idx = tid; idx < 256; idx += 32) {
    int i = idx >> 4, j = idx & 15;
    Ad[(b * K_ + i) * K_ + j] = M[i][16 + j];
  }
}

// ---------------- fused phi + ck via WMMA; one 16-pixel tile per wave ----------------
__global__ void __launch_bounds__(256) phi_ck(const _Float16* __restrict__ gp,
                                              const float* __restrict__ gsq,
                                              const float* __restrict__ centers,
                                              const float* __restrict__ Ad,
                                              _Float16* __restrict__ phi,
                                              _Float16* __restrict__ ck) {
  __shared__ _Float16 tile[8][16][17];               // per-wave phi tile (C-layout -> A-layout relay)
  __shared__ float csq_l[K_];
  int tid  = threadIdx.x;
  int w    = tid >> 5;
  int lane = tid & 31;
  int tileId = blockIdx.x * 8 + w;                   // 16384 tiles per batch
  int b    = tileId >> 14;
  int t    = tileId & 16383;
  int pix0 = t << 4;
  if (tid < K_) {
    float s = 0.f;
#pragma unroll
    for (int f = 0; f < F_; ++f) { float c = centers[(b * K_ + tid) * F_ + f]; s += c * c; }
    csq_l[tid] = s;
  }
  __syncthreads();

  // A: 16 pixels x 8 features (K=0..7 live on lanes<16, elements 0..7); rest zero.
  v16h a = {};
  v16h bm = {};
  if (lane < 16) {
    union { uint4 u; _Float16 h[8]; } cv;
    cv.u = *reinterpret_cast<const uint4*>(gp + ((size_t)b * N_ + pix0 + lane) * 8);
#pragma unroll
    for (int f = 0; f < F_; ++f) a[f] = cv.h[f];
#pragma unroll
    for (int f = 0; f < F_; ++f) bm[f] = (_Float16)centers[(b * K_ + lane) * F_ + f];
  }
  v8f acc = {};
  acc = __builtin_amdgcn_wmma_f32_16x16x32_f16(false, a, false, bm, (short)0, acc, false, false);

  int n  = lane & 15;
  int hi = (lane >> 4) & 1;
  float csq = csq_l[n];
  union { uint4 u; _Float16 h[8]; } st;
#pragma unroll
  for (int i = 0; i < 8; ++i) {
    int m = i + hi * 8;
    float gs = gsq[(size_t)b * N_ + pix0 + m];
    float sq = gs - 2.f * acc[i] + csq;
    float ph = __expf(-sq * 0.25f);
    _Float16 phh = (_Float16)ph;
    st.h[i] = phh;
    tile[w][m][n] = phh;
  }
  *reinterpret_cast<uint4*>(phi + ((size_t)(b * K_ + n)) * N_ + pix0 + hi * 8) = st.u;
  __syncthreads();

  // ck tile = phi(16x16) x Ad^T(16x16), K padded to 32 with zeros
  v16h a2 = {};
  v16h b2 = {};
  {
    int m  = lane & 15;
    int kb = hi * 8;
#pragma unroll
    for (int e = 0; e < 8; ++e) a2[e] = tile[w][m][kb + e];
  }
  {
#pragma unroll
    for (int e = 0; e < 8; ++e) b2[e] = (_Float16)Ad[(b * K_ + hi * 8 + e) * K_ + (lane & 15)];  // Ad symmetric
  }
  v8f acc2 = {};
  acc2 = __builtin_amdgcn_wmma_f32_16x16x32_f16(false, a2, false, b2, (short)0, acc2, false, false);
  union { uint4 u; _Float16 h[8]; } st2;
#pragma unroll
  for (int i = 0; i < 8; ++i) st2.h[i] = (_Float16)acc2[i];
  *reinterpret_cast<uint4*>(ck + ((size_t)(b * K_ + n)) * N_ + pix0 + hi * 8) = st2.u;
}

// ---------------- horizontal blur of phi and phi*input (zero padded) ----------------
__global__ void __launch_bounds__(256) hblur(const _Float16* __restrict__ phi,
                                             const float* __restrict__ inp,
                                             _Float16* __restrict__ tA,
                                             _Float16* __restrict__ tB) {
  __shared__ float sph[256 + 2 * RAD];
  __shared__ float spr[256 + 2 * RAD];
  int gid = blockIdx.x;                              // plane(64) * y(512) * chunk(2)
  int xc = gid & 1;
  int y  = (gid >> 1) & (H_ - 1);
  int p  = gid >> 10;
  int b  = p >> 4;
  int x0 = xc * 256;
  const _Float16* prow = phi + (size_t)p * N_ + y * W_;
  const float*    irow = inp + (size_t)b * N_ + y * W_;
  for (int i = threadIdx.x; i < 256 + 2 * RAD; i += 256) {
    int gx = x0 + i - RAD;
    float pv = 0.f, iv = 0.f;
    if (gx >= 0 && gx < W_) { pv = (float)prow[gx]; iv = irow[gx]; }
    sph[i] = pv;
    spr[i] = pv * iv;
  }
  __syncthreads();
  float av = 0.f, bv = 0.f;
#pragma unroll
  for (int d = 0; d < KS; ++d) {
    float wv = SW[d];
    av += wv * sph[threadIdx.x + d];
    bv += wv * spr[threadIdx.x + d];
  }
  size_t o = (size_t)p * N_ + y * W_ + x0 + threadIdx.x;
  tA[o] = (_Float16)av;
  tB[o] = (_Float16)bv;
}

// ---------------- vertical blur + ck-weighted combine + divide ----------------
__global__ void __launch_bounds__(256) vcombine(const _Float16* __restrict__ tA,
                                                const _Float16* __restrict__ tB,
                                                const _Float16* __restrict__ ck,
                                                float* __restrict__ out) {
  int gid = blockIdx.x * 256 + threadIdx.x;          // 0 .. B*H*W-1
  int x = gid & (W_ - 1);
  int y = (gid >> 9) & (H_ - 1);
  int b = gid >> 18;
  float num = 0.f, den = 0.f;
#pragma unroll 1
  for (int k = 0; k < K_; ++k) {
    const size_t pb = ((size_t)(b * K_ + k)) * N_;
    if (k + 1 < K_) __builtin_prefetch(tA + pb + N_ + (size_t)y * W_ + x, 0, 1);
    float ckv = (float)ck[pb + (size_t)y * W_ + x];
    float ra = 0.f, rb = 0.f;
#pragma unroll
    for (int d = 0; d < KS; ++d) {
      int yy = y + d - RAD;
      if (yy >= 0 && yy < H_) {
        float wv = SW[d];
        ra += wv * (float)tA[pb + (size_t)yy * W_ + x];
        rb += wv * (float)tB[pb + (size_t)yy * W_ + x];
      }
    }
    den += ckv * ra;
    num += ckv * rb;
  }
  out[gid] = num / den;
}

extern "C" void kernel_launch(void* const* d_in, const int* in_sizes, int n_in,
                              void* d_out, int out_size, void* d_ws, size_t ws_size,
                              hipStream_t stream) {
  (void)in_sizes; (void)n_in; (void)out_size; (void)ws_size;
  const float* input = (const float*)d_in[0];       // (4,1,512,512) f32
  const float* guide = (const float*)d_in[1];       // (4,8,512,512) f32

  char* ws = (char*)d_ws;
  size_t off = 0;
  auto alloc = [&](size_t bytes) -> char* {
    char* p = ws + off;
    off += (bytes + 255) & ~(size_t)255;
    return p;
  };
  _Float16* gp   = (_Float16*)alloc((size_t)B_ * N_ * 8 * 2);       // 16.8 MB
  float*    gsq  = (float*)   alloc((size_t)B_ * N_ * 4);           //  4.2 MB
  _Float16* phi  = (_Float16*)alloc((size_t)B_ * K_ * N_ * 2);      // 33.6 MB
  _Float16* ckb  = (_Float16*)alloc((size_t)B_ * K_ * N_ * 2);      // 33.6 MB
  _Float16* tA   = (_Float16*)alloc((size_t)B_ * K_ * N_ * 2);      // 33.6 MB
  _Float16* tB   = (_Float16*)alloc((size_t)B_ * K_ * N_ * 2);      // 33.6 MB
  float* centers = (float*)alloc(B_ * K_ * F_ * 4);
  float* sums    = (float*)alloc((B_ * K_ * F_ + B_ * K_) * 4);     // sums | counts contiguous
  float* counts  = sums + B_ * K_ * F_;
  float* Ad      = (float*)alloc(B_ * K_ * K_ * 4);

  pack_guide<<<(B_ * N_) / 256, 256, 0, stream>>>(guide, gp, gsq);
  init_centers<<<B_, 128, 0, stream>>>(gp, centers);
  for (int it = 0; it < 10; ++it) {
    hipMemsetAsync(sums, 0, (size_t)(B_ * K_ * F_ + B_ * K_) * 4, stream);
    assign_accum<<<B_ * 128, 256, 0, stream>>>(gp, centers, sums, counts);
    update_centers<<<B_, 128, 0, stream>>>(centers, sums, counts);
  }
  pinv16<<<B_, 32, 0, stream>>>(centers, Ad);
  phi_ck<<<(B_ * (N_ / 16)) / 8, 256, 0, stream>>>(gp, gsq, centers, Ad, phi, ckb);
  hblur<<<B_ * K_ * H_ * 2, 256, 0, stream>>>(phi, input, tA, tB);
  vcombine<<<(B_ * N_) / 256, 256, 0, stream>>>(tA, tB, ckb, (float*)d_out);
}
